// VQFFN1_3702261809276
// MI455X (gfx1250) — matrix operand order repaired and checked by reading, not compiled, not run on hardware
//
#include <hip/hip_runtime.h>

typedef __bf16 bf16;
typedef __attribute__((ext_vector_type(16))) __bf16 v16bf;
typedef __attribute__((ext_vector_type(8)))  __bf16 bf16x8;
typedef __attribute__((ext_vector_type(8)))  float  v8f;
typedef __attribute__((ext_vector_type(4)))  float  f32x4;

#define D_MODEL 1024
#define D_FF    4096
#define EPS     1e-6f

#define ROWS     32            // M rows per workgroup (2 WMMA M-tiles)
#define NWAVES   8             // 256 threads, wave32; wave pairs share B lines
#define AS_PITCH (D_MODEL + 8) // 1032 bf16 -> 2064B rows (16B aligned)
#define CHUNK    256           // D_FF columns per phase-2 chunk
#define PB_PITCH (CHUNK + 8)   // 264 bf16 -> 528B rows (16B aligned)
#define PIPE     4             // register B pipeline depth (phase 2)
#define SB_SLOTS 4             // async LDS staging slots per wave (phase 1)
#define SB_SLOT_B 1024         // bytes per slot: 32 lanes x 32B
#define NEG_INF  (-1e30f)

// ---------------------------------------------------------------------------
// Kernel 1: wn[c][k] = rmsnorm(w[c])[k] as bf16 (B operand of GEMM1, K-contig)
// ---------------------------------------------------------------------------
__global__ __launch_bounds__(256) void prep_wn(const float* __restrict__ w,
                                               bf16* __restrict__ wn) {
  __shared__ float red[8];
  const int c = blockIdx.x;
  const int t = threadIdx.x;
  const f32x4* wrow = (const f32x4*)(w + (size_t)c * D_MODEL);
  f32x4 v = wrow[t];
  float ss = v.x * v.x + v.y * v.y + v.z * v.z + v.w * v.w;
  #pragma unroll
  for (int m = 16; m >= 1; m >>= 1) ss += __shfl_xor(ss, m, 32);
  if ((t & 31) == 0) red[t >> 5] = ss;
  __syncthreads();
  float tot = 0.f;
  #pragma unroll
  for (int i = 0; i < 8; ++i) tot += red[i];
  const float r = rsqrtf(tot * (1.0f / D_MODEL) + EPS);
  bf16* o = wn + (size_t)c * D_MODEL + (size_t)t * 4;
  o[0] = (bf16)(v.x * r); o[1] = (bf16)(v.y * r);
  o[2] = (bf16)(v.z * r); o[3] = (bf16)(v.w * r);
}

// ---------------------------------------------------------------------------
// Kernel 2: wT[d][c] = bf16(w[c][d])  (B operand of GEMM2, K(=c)-contiguous)
// ---------------------------------------------------------------------------
__global__ __launch_bounds__(256) void prep_wT(const float* __restrict__ w,
                                               bf16* __restrict__ wT) {
  __shared__ bf16 tile[32][33];
  const int d0 = blockIdx.x * 32;
  const int c0 = blockIdx.y * 32;
  const int tx = threadIdx.x, ty = threadIdx.y;   // 32 x 8
  #pragma unroll
  for (int i = 0; i < 4; ++i) {
    int c = c0 + ty + i * 8;
    tile[ty + i * 8][tx] = (bf16)w[(size_t)c * D_MODEL + d0 + tx];
  }
  __syncthreads();
  #pragma unroll
  for (int i = 0; i < 4; ++i) {
    int d = d0 + ty + i * 8;
    wT[(size_t)d * D_FF + c0 + tx] = tile[tx][ty + i * 8];
  }
}

// ---------------------------------------------------------------------------
// Fused: rmsnorm(x) -> WMMA scores (async-LDS B stream) -> online softmax ->
//        per-chunk recompute -> bf16 probs (LDS) -> WMMA GEMM2 (reg pipeline)
// ---------------------------------------------------------------------------
__global__ __launch_bounds__(256) void vqffn_fused(
    const float* __restrict__ x,   // [16384][1024]
    const bf16* __restrict__ wn,   // [4096][1024]
    const bf16* __restrict__ wT,   // [1024][4096]
    float* __restrict__ out) {     // [16384][1024]

  __shared__ __align__(16) bf16 As[ROWS][AS_PITCH];                 // ~66 KB
  __shared__ __align__(16) bf16 Pb[ROWS][PB_PITCH];                 // ~17 KB
  __shared__ __align__(16) bf16 SB[NWAVES][SB_SLOTS * SB_SLOT_B / 2]; // 32 KB
  __shared__ float grp_m[4][ROWS];
  __shared__ float grp_s[4][ROWS];
  __shared__ float row_m[ROWS];
  __shared__ float row_is[ROWS];

  const int tid  = threadIdx.x;
  const int wave = tid >> 5;
  const int lane = tid & 31;
  const int half = lane >> 4;
  const int ln   = lane & 15;
  const int mt   = wave & 1;        // which 16-row M-tile this wave computes
  const int wg   = wave >> 1;       // column group 0..3
  const size_t row_base = (size_t)blockIdx.x * ROWS;

  // ---- Phase 0: RMSNorm 32 rows of x into LDS (bf16), NT streaming loads --
  {
    const int r8  = tid >> 4;
    const int seg = tid & 15;
    #pragma unroll 1
    for (int rr = 0; rr < ROWS; rr += 16) {
      const int r = rr + r8;
      const f32x4* xr = (const f32x4*)(x + (row_base + r) * D_MODEL);
      f32x4 v[16];
      float ss = 0.f;
      #pragma unroll
      for (int j = 0; j < 16; ++j) {
        v[j] = __builtin_nontemporal_load(xr + seg + j * 16);
        ss += v[j].x * v[j].x + v[j].y * v[j].y + v[j].z * v[j].z + v[j].w * v[j].w;
      }
      #pragma unroll
      for (int m = 8; m >= 1; m >>= 1) ss += __shfl_xor(ss, m, 32);
      const float rs = rsqrtf(ss * (1.0f / D_MODEL) + EPS);
      #pragma unroll
      for (int j = 0; j < 16; ++j) {
        const int kb = (seg + j * 16) * 4;
        As[r][kb + 0] = (bf16)(v[j].x * rs);
        As[r][kb + 1] = (bf16)(v[j].y * rs);
        As[r][kb + 2] = (bf16)(v[j].z * rs);
        As[r][kb + 3] = (bf16)(v[j].w * rs);
      }
    }
  }
  __syncthreads();

  const bf16* SBw = &SB[wave][0];
  const uint32_t sb_lane = (uint32_t)(uintptr_t)&SB[wave][0] + (uint32_t)lane * 32u;

  // -------- Phase-1 score tile: B streamed global->LDS with ASYNC copies ---
  // 4 slots x 1KB per wave; ASYNCcnt completes in order, so waiting <=6
  // guarantees the oldest slot has landed in LDS. VGPR cost for B: none
  // while in flight.
  auto score_tile_async = [&](int c0) -> v8f {
    v8f c = {};
    const bf16* bp = wn + (size_t)(c0 + ln) * D_MODEL + (size_t)(half * 16);
    const int ar = mt * 16 + ln;
    #pragma unroll
    for (int p = 0; p < SB_SLOTS - 1; ++p) {
      asm volatile("global_load_async_to_lds_b128 %0, %1, off"
                   :: "v"(sb_lane + (uint32_t)(p * SB_SLOT_B)),
                      "v"((const void*)(bp + p * 32)) : "memory");
      asm volatile("global_load_async_to_lds_b128 %0, %1, off"
                   :: "v"(sb_lane + (uint32_t)(p * SB_SLOT_B + 16)),
                      "v"((const void*)(bp + p * 32 + 8)) : "memory");
    }
    #pragma unroll
    for (int it = 0; it < 28; ++it) {
      const int kb = it * 32;
      const int ps = (it + SB_SLOTS - 1) & (SB_SLOTS - 1);
      asm volatile("global_load_async_to_lds_b128 %0, %1, off"
                   :: "v"(sb_lane + (uint32_t)(ps * SB_SLOT_B)),
                      "v"((const void*)(bp + kb + 96)) : "memory");
      asm volatile("global_load_async_to_lds_b128 %0, %1, off"
                   :: "v"(sb_lane + (uint32_t)(ps * SB_SLOT_B + 16)),
                      "v"((const void*)(bp + kb + 96 + 8)) : "memory");
      asm volatile("s_wait_asynccnt 0x6" ::: "memory");
      const bf16* sl = SBw + (size_t)((it & (SB_SLOTS - 1)) * 512 + lane * 16);
      bf16x8 b0 = *(const bf16x8*)(sl);
      bf16x8 b1 = *(const bf16x8*)(sl + 8);
      bf16x8 a0 = *(const bf16x8*)(&As[ar][half * 16 + kb]);
      bf16x8 a1 = *(const bf16x8*)(&As[ar][half * 16 + kb + 8]);
      v16bf a, b;
      #pragma unroll
      for (int i = 0; i < 8; ++i) {
        a[i] = a0[i]; a[i + 8] = a1[i];
        b[i] = b0[i]; b[i + 8] = b1[i];
      }
      c = __builtin_amdgcn_wmma_f32_16x16x32_bf16(
              false, a, false, b, (short)0, c, false, false);
    }
    asm volatile("s_wait_asynccnt 0x0" ::: "memory");
    #pragma unroll
    for (int it = 28; it < 32; ++it) {       // drain last 4 slots
      const int kb = it * 32;
      const bf16* sl = SBw + (size_t)((it & (SB_SLOTS - 1)) * 512 + lane * 16);
      bf16x8 b0 = *(const bf16x8*)(sl);
      bf16x8 b1 = *(const bf16x8*)(sl + 8);
      bf16x8 a0 = *(const bf16x8*)(&As[ar][half * 16 + kb]);
      bf16x8 a1 = *(const bf16x8*)(&As[ar][half * 16 + kb + 8]);
      v16bf a, b;
      #pragma unroll
      for (int i = 0; i < 8; ++i) {
        a[i] = a0[i]; a[i + 8] = a1[i];
        b[i] = b0[i]; b[i + 8] = b1[i];
      }
      c = __builtin_amdgcn_wmma_f32_16x16x32_bf16(
              false, a, false, b, (short)0, c, false, false);
    }
    return c;
  };

  // -------- Phase-2 score tile: register B pipeline, pinned with
  //          sched_barrier(0) so the scheduler cannot re-serialize it. ------
  auto score_tile_reg = [&](int c0) -> v8f {
    v8f c = {};
    const bf16* bp = wn + (size_t)(c0 + ln) * D_MODEL + (size_t)(half * 16);
    const int ar = mt * 16 + ln;
    bf16x8 pb0[PIPE], pb1[PIPE];
    #pragma unroll
    for (int p = 0; p < PIPE - 1; ++p) {
      pb0[p] = *(const bf16x8*)(bp + p * 32);
      pb1[p] = *(const bf16x8*)(bp + p * 32 + 8);
    }
    #pragma unroll
    for (int it = 0; it < D_MODEL / 32; ++it) {
      const int kb = it * 32;
      const int pk = kb + (PIPE - 1) * 32;
      if (pk < D_MODEL) {
        pb0[(it + PIPE - 1) % PIPE] = *(const bf16x8*)(bp + pk);
        pb1[(it + PIPE - 1) % PIPE] = *(const bf16x8*)(bp + pk + 8);
      }
      bf16x8 a0 = *(const bf16x8*)(&As[ar][half * 16 + kb]);
      bf16x8 a1 = *(const bf16x8*)(&As[ar][half * 16 + kb + 8]);
      __builtin_amdgcn_sched_barrier(0);   // keep prefetches above the WMMA
      v16bf a, b;
      #pragma unroll
      for (int i = 0; i < 8; ++i) {
        a[i] = a0[i]; a[i + 8] = a1[i];
        b[i] = pb0[it % PIPE][i]; b[i + 8] = pb1[it % PIPE][i];
      }
      c = __builtin_amdgcn_wmma_f32_16x16x32_bf16(
              false, a, false, b, (short)0, c, false, false);
    }
    return c;
  };

  // ---- Phase 1: online softmax stats; wave covers 1024 cols of its M-tile -
  float m_run[8], s_run[8];
  #pragma unroll
  for (int v = 0; v < 8; ++v) { m_run[v] = NEG_INF; s_run[v] = 0.f; }

  #pragma unroll 1
  for (int t = 0; t < 64; ++t) {
    const int c0 = wg * 1024 + t * 16;
    v8f c = score_tile_async(c0);
    #pragma unroll
    for (int v = 0; v < 8; ++v) {
      float tmax = c[v];
      #pragma unroll
      for (int m = 8; m >= 1; m >>= 1) tmax = fmaxf(tmax, __shfl_xor(tmax, m, 32));
      const float nm = fmaxf(m_run[v], tmax);
      float e = __expf(c[v] - nm);
      #pragma unroll
      for (int m = 8; m >= 1; m >>= 1) e += __shfl_xor(e, m, 32);
      s_run[v] = s_run[v] * __expf(m_run[v] - nm) + e;
      m_run[v] = nm;
    }
  }
  if (ln == 0) {
    #pragma unroll
    for (int v = 0; v < 8; ++v) {
      const int r = mt * 16 + v + 8 * half;
      grp_m[wg][r] = m_run[v];
      grp_s[wg][r] = s_run[v];
    }
  }
  __syncthreads();
  if (tid < ROWS) {
    float m = NEG_INF;
    #pragma unroll
    for (int g = 0; g < 4; ++g) m = fmaxf(m, grp_m[g][tid]);
    float s = 0.f;
    #pragma unroll
    for (int g = 0; g < 4; ++g) s += grp_s[g][tid] * __expf(grp_m[g][tid] - m);
    row_m[tid]  = m;
    row_is[tid] = 1.f / s;
  }
  __syncthreads();

  // ---- Phase 2: chunks of 256 cols: recompute -> probs -> GEMM2 ----------
  v8f acc[16];                       // 16 N-tiles x 8 f32 = 128 VGPRs
  #pragma unroll
  for (int nt = 0; nt < 16; ++nt) acc[nt] = (v8f){};

  #pragma unroll 1
  for (int ch = 0; ch < D_FF; ch += CHUNK) {
    // 2a: wave recomputes 4 column-tiles of its M-tile -> probs into Pb
    #pragma unroll 1
    for (int t = 0; t < 4; ++t) {
      const int cl0 = (wg * 4 + t) * 16;
      v8f c = score_tile_reg(ch + cl0);
      #pragma unroll
      for (int v = 0; v < 8; ++v) {
        const int r = mt * 16 + v + 8 * half;
        const float p = __expf(c[v] - row_m[r]) * row_is[r];
        Pb[r][cl0 + ln] = (bf16)p;   // C-layout -> A-layout via LDS
      }
    }
    __syncthreads();

    // 2b: wave accumulates its (M-tile, 256 d-cols) over this K chunk.
    #pragma unroll
    for (int nt = 0; nt < 16; ++nt) {
      const int d0 = wg * 256 + nt * 16;
      const bf16* bp = wT + (size_t)(d0 + ln) * D_FF + ch + (size_t)(half * 16);
      bf16x8 pb0[PIPE], pb1[PIPE];
      #pragma unroll
      for (int p = 0; p < PIPE - 1; ++p) {
        pb0[p] = *(const bf16x8*)(bp + p * 32);
        pb1[p] = *(const bf16x8*)(bp + p * 32 + 8);
      }
      v8f c = acc[nt];
      #pragma unroll
      for (int it = 0; it < CHUNK / 32; ++it) {   // 8 K-steps
        const int kk = it * 32;
        const int pk = kk + (PIPE - 1) * 32;
        if (pk < CHUNK) {
          pb0[(it + PIPE - 1) % PIPE] = *(const bf16x8*)(bp + pk);
          pb1[(it + PIPE - 1) % PIPE] = *(const bf16x8*)(bp + pk + 8);
        }
        bf16x8 a0 = *(const bf16x8*)(&Pb[mt * 16 + ln][half * 16 + kk]);
        bf16x8 a1 = *(const bf16x8*)(&Pb[mt * 16 + ln][half * 16 + kk + 8]);
        __builtin_amdgcn_sched_barrier(0);   // pin prefetches above the WMMA
        v16bf a, b;
        #pragma unroll
        for (int i = 0; i < 8; ++i) {
          a[i] = a0[i]; a[i + 8] = a1[i];
          b[i] = pb0[it % PIPE][i]; b[i + 8] = pb1[it % PIPE][i];
        }
        c = __builtin_amdgcn_wmma_f32_16x16x32_bf16(
                false, a, false, b, (short)0, c, false, false);
      }
      acc[nt] = c;
    }
    __syncthreads();   // protect Pb before next chunk overwrites it
  }

  // ---- Epilogue: NT-stream f32 output (never re-read) --------------------
  #pragma unroll
  for (int nt = 0; nt < 16; ++nt) {
    const int d0 = wg * 256 + nt * 16;
    #pragma unroll
    for (int v = 0; v < 8; ++v) {
      const int r = mt * 16 + v + 8 * half;
      __builtin_nontemporal_store(acc[nt][v],
          &out[(row_base + r) * D_MODEL + d0 + ln]);
    }
  }
}

// ---------------------------------------------------------------------------
extern "C" void kernel_launch(void* const* d_in, const int* in_sizes, int n_in,
                              void* d_out, int out_size, void* d_ws, size_t ws_size,
                              hipStream_t stream) {
  (void)in_sizes; (void)n_in; (void)out_size; (void)ws_size;
  const float* x = (const float*)d_in[0];   // [4,4096,1024] f32
  const float* w = (const float*)d_in[1];   // [4096,1024]   f32

  bf16* wn = (bf16*)d_ws;                              // 8 MB
  bf16* wT = wn + (size_t)D_FF * D_MODEL;              // 8 MB

  prep_wn<<<D_FF, 256, 0, stream>>>(w, wn);
  dim3 tb(32, 8);
  dim3 tg(D_MODEL / 32, D_FF / 32);
  prep_wT<<<tg, tb, 0, stream>>>(w, wT);

  const int n_rows = 4 * 4096;                         // 16384
  vqffn_fused<<<n_rows / ROWS, 256, 0, stream>>>(x, wn, wT, (float*)d_out);
}
